// Seq2SeqRNN_66254165508284
// MI455X (gfx1250) — compile-verified, matrix operand order checked by reading, not run on hardware
//
#include <hip/hip_runtime.h>
#include <hip/hip_bf16.h>

// Problem constants from the reference
#define BB 16
#define SS 128
#define TT 64
#define VV 32000
#define EE 256
#define HH 256

typedef __attribute__((ext_vector_type(16))) __bf16 v16bf;
typedef __attribute__((ext_vector_type(8)))  float  v8f;

// ---------------------------------------------------------------------------
// Helpers
// ---------------------------------------------------------------------------
__device__ __forceinline__ __bf16 f2bf(float f) {
    union { float f; unsigned u; } v; v.f = f;
    unsigned r = (v.u + 0x7FFFu + ((v.u >> 16) & 1u)) >> 16;
    union { unsigned short s; __bf16 b; } o; o.s = (unsigned short)r;
    return o.b;
}

__device__ __forceinline__ float sigmoidf(float x) {
    return 1.0f / (1.0f + __expf(-x));
}

// ISA 7.12.2: 16-bit A/B operand K index for (element e, lane) within a 32-K tile.
__device__ __forceinline__ int kidx16(int e, int lane) {
    return e + (e >= 8 ? 8 : 0) + (lane >= 16 ? 8 : 0);
}

// Inverse mapping: flat index inside a [8][32][16] A-layout block for (row m, k).
__device__ __forceinline__ int a_index(int m, int k) {
    int kt = k >> 5, kk = k & 31;
    int laneoff = ((kk >> 3) & 1) ? 16 : 0;
    int e = (kk & 7) + ((kk >= 16) ? 8 : 0);
    return kt * 512 + (laneoff + m) * 16 + e;
}

__device__ __forceinline__ v8f wmma_bf16(v16bf a, v16bf b, v8f c) {
    return __builtin_amdgcn_wmma_f32_16x16x32_bf16(
        /*neg_a=*/false, a, /*neg_b=*/false, b,
        /*c_mod=*/(short)0, c, /*reuse_a=*/false, /*reuse_b=*/false);
}

// Compiler fence: forbid hoisting (LICM) of loads across loop iterations so the
// recurrent kernels re-read L0/L2-resident weight panels instead of spilling
// 512 VGPRs worth of "invariant" operands to scratch.
__device__ __forceinline__ void no_hoist() { asm volatile("" ::: "memory"); }

// ---------------------------------------------------------------------------
// Weight repack: W (N x K, row-major fp32, used as x @ W^T) -> bf16 WMMA B panels
// layout: [nt][kt][lane(32)][e(16)]
// ---------------------------------------------------------------------------
__global__ void conv_b_panel(const float* __restrict__ W, __bf16* __restrict__ out,
                             int N, int K) {
    int idx = blockIdx.x * blockDim.x + threadIdx.x;
    int total = N * K;
    if (idx >= total) return;
    int e = idx & 15;
    int lane = (idx >> 4) & 31;
    int tile = idx >> 9;
    int KT = K >> 5;
    int kt = tile % KT, nt = tile / KT;
    int n = nt * 16 + (lane & 15);
    int k = kt * 32 + kidx16(e, lane);
    out[idx] = f2bf(W[(size_t)n * K + k]);
}

// ---------------------------------------------------------------------------
// Embedding gather straight into per-timestep WMMA A-layout tiles:
// out[t][kt(8)][lane(32)][e(16)], rows m = batch, K = embedding dim (256).
// ---------------------------------------------------------------------------
__global__ void gather_A(const int* __restrict__ seq, int seq_stride, int nsteps,
                         const float* __restrict__ emb, __bf16* __restrict__ out) {
    int idx = blockIdx.x * blockDim.x + threadIdx.x;
    int total = nsteps * 4096;
    if (idx >= total) return;
    int t = idx >> 12;
    int r = idx & 4095;
    int e = r & 15, lane = (r >> 4) & 31, kt = r >> 9;
    int b = lane & 15;
    int k = kt * 32 + kidx16(e, lane);
    int token = seq[b * seq_stride + t];
    out[idx] = f2bf(emb[(size_t)token * EE + k]);
}

// ---------------------------------------------------------------------------
// Encoder LSTM: one persistent workgroup, 16 waves. Wave w owns the i/f/g/o
// gate tiles for hidden block [16w,16w+16); nonlinearity stays in registers.
// kt-outer / gate-inner loop order: each A fragment feeds 4 WMMAs.
// ---------------------------------------------------------------------------
__global__ void __launch_bounds__(512)
encoder_lstm(const __bf16* __restrict__ srcA,
             const __bf16* __restrict__ WihB, const __bf16* __restrict__ WhhB,
             const float* __restrict__ b4,
             __bf16* __restrict__ enc_hs_A,
             __bf16* __restrict__ h_state_A, float* __restrict__ c_state) {
    __shared__ __attribute__((aligned(32))) __bf16 hA[4096];  // h in A-layout
    __shared__ float c_s[4096];                               // c, [b][hid]
    int tid = threadIdx.x;
    for (int i = tid; i < 4096; i += 512) { hA[i] = f2bf(0.0f); c_s[i] = 0.0f; }
    __syncthreads();

    int w = tid >> 5, lane = tid & 31;
    int nl = lane & 15;
    int moff = (lane >= 16) ? 8 : 0;
    int hid = w * 16 + nl;

    for (int t = 0; t < SS; ++t) {
        no_hoist();  // weights must be re-fetched (L0/L2 hit), not scratch-spilled
        v8f acc[4] = {v8f{}, v8f{}, v8f{}, v8f{}};
        const __bf16* ap = srcA + (size_t)t * 4096;
        #pragma unroll
        for (int kt = 0; kt < 8; ++kt) {
            v16bf a = *(const v16bf*)(ap + kt * 512 + lane * 16);
            #pragma unroll
            for (int g = 0; g < 4; ++g) {
                int nt = g * 16 + w;
                acc[g] = wmma_bf16(a,
                    *(const v16bf*)(WihB + (size_t)(nt * 8 + kt) * 512 + lane * 16),
                    acc[g]);
            }
        }
        #pragma unroll
        for (int kt = 0; kt < 8; ++kt) {
            v16bf a = *(const v16bf*)&hA[kt * 512 + lane * 16];
            #pragma unroll
            for (int g = 0; g < 4; ++g) {
                int nt = g * 16 + w;
                acc[g] = wmma_bf16(a,
                    *(const v16bf*)(WhhB + (size_t)(nt * 8 + kt) * 512 + lane * 16),
                    acc[g]);
            }
        }
        float bi_i = b4[0 * 256 + hid], bi_f = b4[1 * 256 + hid];
        float bi_g = b4[2 * 256 + hid], bi_o = b4[3 * 256 + hid];
        __syncthreads();  // all waves done reading hA for this step
        #pragma unroll
        for (int r = 0; r < 8; ++r) {
            int m = r + moff;
            float ig = sigmoidf(acc[0][r] + bi_i);
            float fg = sigmoidf(acc[1][r] + bi_f);
            float gg = tanhf(acc[2][r] + bi_g);
            float og = sigmoidf(acc[3][r] + bi_o);
            float cn = fg * c_s[m * 256 + hid] + ig * gg;
            c_s[m * 256 + hid] = cn;
            float h = og * tanhf(cn);
            int ai = a_index(m, hid);
            __bf16 hb = f2bf(h);
            hA[ai] = hb;
            enc_hs_A[(size_t)t * 4096 + ai] = hb;  // feeds K/V GEMM
        }
        __syncthreads();
    }
    for (int i = tid; i < 4096; i += 512) { h_state_A[i] = hA[i]; c_state[i] = c_s[i]; }
}

// ---------------------------------------------------------------------------
// K = enc_out @ Wk^T + bk, Vp = enc_out @ Wv^T + bv. One wave per (s, nt, K|V).
// ---------------------------------------------------------------------------
__global__ void __launch_bounds__(1024)
kv_gemm(const __bf16* __restrict__ enc_hs_A,
        const __bf16* __restrict__ WkB, const __bf16* __restrict__ WvB,
        const float* __restrict__ bk, const float* __restrict__ bv,
        float* __restrict__ Kf, float* __restrict__ Vf) {
    int s = blockIdx.x;                 // timestep 0..127 (M-tile rows = batch)
    int w = threadIdx.x >> 5, lane = threadIdx.x & 31;
    int which = w >> 4;                 // 0 -> K, 1 -> V
    int nt = w & 15;
    const __bf16* Bm = which ? WvB : WkB;
    const float*  bias = which ? bv : bk;
    float* out = which ? Vf : Kf;
    v8f c = {};
    const __bf16* ap = enc_hs_A + (size_t)s * 4096;
    #pragma unroll
    for (int kt = 0; kt < 8; ++kt)
        c = wmma_bf16(*(const v16bf*)(ap + kt * 512 + lane * 16),
                      *(const v16bf*)(Bm + (size_t)(nt * 8 + kt) * 512 + lane * 16), c);
    int nl = lane & 15, moff = (lane >= 16) ? 8 : 0;
    float bi = bias[nt * 16 + nl];
    #pragma unroll
    for (int r = 0; r < 8; ++r) {
        int b = r + moff;
        out[(size_t)b * SS * HH + (size_t)s * HH + nt * 16 + nl] = c[r] + bi;
    }
}

// ---------------------------------------------------------------------------
// Decoder step: q GEMM (WMMA) -> scores -> softmax -> ctx -> Wo GEMM (WMMA)
// -> gate GEMM (WMMA, K=512 input + K=256 recurrent) -> LSTM update.
// ---------------------------------------------------------------------------
__global__ void __launch_bounds__(512)
dec_step(int t,
         const __bf16* __restrict__ tokA,
         const __bf16* __restrict__ WqB, const float* __restrict__ bq,
         const float* __restrict__ Kf, const float* __restrict__ Vf,
         const __bf16* __restrict__ WoB, const float* __restrict__ bo,
         const __bf16* __restrict__ WihB, const __bf16* __restrict__ WhhB,
         const float* __restrict__ b4,
         __bf16* __restrict__ h_state_A, float* __restrict__ c_state) {
    __shared__ __attribute__((aligned(32))) __bf16 hA[4096];
    __shared__ float c_s[4096];
    __shared__ float q_s[4096];     // q, later reused for ctx2 staging
    __shared__ float sc[2048];      // scores -> attn weights
    __shared__ __attribute__((aligned(32))) __bf16 ctxA[4096];
    __shared__ __attribute__((aligned(32))) __bf16 ctx2A[4096];

    int tid = threadIdx.x;
    for (int i = tid; i < 4096; i += 512) { hA[i] = h_state_A[i]; c_s[i] = c_state[i]; }
    __syncthreads();

    int w = tid >> 5, lane = tid & 31;
    int nl = lane & 15, moff = (lane >= 16) ? 8 : 0;

    // --- q = h @ Wq^T + bq (wave w -> N-tile w) ---
    {
        v8f c = {};
        #pragma unroll
        for (int kt = 0; kt < 8; ++kt)
            c = wmma_bf16(*(const v16bf*)&hA[kt * 512 + lane * 16],
                          *(const v16bf*)(WqB + (size_t)(w * 8 + kt) * 512 + lane * 16), c);
        float bi = bq[w * 16 + nl];
        #pragma unroll
        for (int r = 0; r < 8; ++r)
            q_s[(r + moff) * 256 + w * 16 + nl] = c[r] + bi;
    }
    __syncthreads();

    // --- scores[b,s] = (q[b] . K[b,s]) * 1/sqrt(H) ---
    const float scale = 0.0625f;  // 1/sqrt(256)
    for (int p = tid; p < 2048; p += 512) {
        int b = p >> 7, s = p & 127;
        const float* kp = Kf + (size_t)b * SS * HH + (size_t)s * HH;
        float acc = 0.0f;
        for (int h = 0; h < HH; ++h) acc += q_s[b * 256 + h] * kp[h];
        sc[p] = acc * scale;
    }
    __syncthreads();

    // --- softmax per batch row (wave w -> row b=w), wave32 shuffle reduce ---
    {
        int b = w;
        float v0 = sc[b * 128 + lane],      v1 = sc[b * 128 + 32 + lane];
        float v2 = sc[b * 128 + 64 + lane], v3 = sc[b * 128 + 96 + lane];
        float mx = fmaxf(fmaxf(v0, v1), fmaxf(v2, v3));
        for (int off = 16; off > 0; off >>= 1) mx = fmaxf(mx, __shfl_xor(mx, off, 32));
        float e0 = __expf(v0 - mx), e1 = __expf(v1 - mx);
        float e2 = __expf(v2 - mx), e3 = __expf(v3 - mx);
        float sm = e0 + e1 + e2 + e3;
        for (int off = 16; off > 0; off >>= 1) sm += __shfl_xor(sm, off, 32);
        float inv = 1.0f / sm;
        sc[b * 128 + lane] = e0 * inv;      sc[b * 128 + 32 + lane] = e1 * inv;
        sc[b * 128 + 64 + lane] = e2 * inv; sc[b * 128 + 96 + lane] = e3 * inv;
    }
    __syncthreads();

    // --- ctx[b,h] = sum_s attn[b,s] * Vp[b,s,h], written in WMMA A-layout ---
    for (int p = tid; p < 4096; p += 512) {
        int b = p >> 8, hid = p & 255;
        const float* vp = Vf + (size_t)b * SS * HH + hid;
        float acc = 0.0f;
        for (int s = 0; s < SS; ++s) acc += sc[b * 128 + s] * vp[(size_t)s * HH];
        ctxA[a_index(b, hid)] = f2bf(acc);
    }
    __syncthreads();

    // --- ctx2 = ctx @ Wo^T + bo (stage fp32 in q_s) ---
    {
        v8f c = {};
        #pragma unroll
        for (int kt = 0; kt < 8; ++kt)
            c = wmma_bf16(*(const v16bf*)&ctxA[kt * 512 + lane * 16],
                          *(const v16bf*)(WoB + (size_t)(w * 8 + kt) * 512 + lane * 16), c);
        float bi = bo[w * 16 + nl];
        #pragma unroll
        for (int r = 0; r < 8; ++r)
            q_s[(r + moff) * 256 + w * 16 + nl] = c[r] + bi;
    }
    __syncthreads();
    for (int p = tid; p < 4096; p += 512) {
        int b = p >> 8, hid = p & 255;
        ctx2A[a_index(b, hid)] = f2bf(q_s[p]);
    }
    __syncthreads();

    // --- gates = [tok, ctx2] @ Wih^T + h @ Whh^T + b ; wave w owns i/f/g/o ---
    // kt-outer / gate-inner so each A fragment feeds 4 WMMAs.
    v8f acc[4] = {v8f{}, v8f{}, v8f{}, v8f{}};
    const __bf16* xap = tokA + (size_t)t * 4096;
    #pragma unroll
    for (int kt = 0; kt < 8; ++kt) {       // token half of the concat (K 0..255)
        v16bf a = *(const v16bf*)(xap + kt * 512 + lane * 16);
        #pragma unroll
        for (int g = 0; g < 4; ++g) {
            int nt = g * 16 + w;
            acc[g] = wmma_bf16(a,
                *(const v16bf*)(WihB + (size_t)(nt * 16 + kt) * 512 + lane * 16), acc[g]);
        }
    }
    #pragma unroll
    for (int kt = 0; kt < 8; ++kt) {       // ctx half of the concat (K 256..511)
        v16bf a = *(const v16bf*)&ctx2A[kt * 512 + lane * 16];
        #pragma unroll
        for (int g = 0; g < 4; ++g) {
            int nt = g * 16 + w;
            acc[g] = wmma_bf16(a,
                *(const v16bf*)(WihB + (size_t)(nt * 16 + 8 + kt) * 512 + lane * 16), acc[g]);
        }
    }
    #pragma unroll
    for (int kt = 0; kt < 8; ++kt) {       // recurrent term
        v16bf a = *(const v16bf*)&hA[kt * 512 + lane * 16];
        #pragma unroll
        for (int g = 0; g < 4; ++g) {
            int nt = g * 16 + w;
            acc[g] = wmma_bf16(a,
                *(const v16bf*)(WhhB + (size_t)(nt * 8 + kt) * 512 + lane * 16), acc[g]);
        }
    }
    __syncthreads();

    // --- LSTM elementwise update; persist h (A-layout bf16) and c (fp32) ---
    {
        int hid = w * 16 + nl;
        float bi_i = b4[0 * 256 + hid], bi_f = b4[1 * 256 + hid];
        float bi_g = b4[2 * 256 + hid], bi_o = b4[3 * 256 + hid];
        #pragma unroll
        for (int r = 0; r < 8; ++r) {
            int m = r + moff;
            float ig = sigmoidf(acc[0][r] + bi_i);
            float fg = sigmoidf(acc[1][r] + bi_f);
            float gg = tanhf(acc[2][r] + bi_g);
            float og = sigmoidf(acc[3][r] + bi_o);
            float cn = fg * c_s[m * 256 + hid] + ig * gg;
            c_state[m * 256 + hid] = cn;
            float h = og * tanhf(cn);
            h_state_A[a_index(m, hid)] = f2bf(h);
        }
    }
}

// ---------------------------------------------------------------------------
// logits = h @ Wout^T + bout : the dominant GEMM (16 x 32000 x 256).
// 2000 N-tiles, one wave each; Wout streamed from L2 as bf16 panels, with
// a software prefetch running one K-panel ahead of the WMMA consumption.
// ---------------------------------------------------------------------------
__global__ void __launch_bounds__(512)
logits_gemm(int t, const __bf16* __restrict__ hA,
            const __bf16* __restrict__ WoutB, const float* __restrict__ bout,
            float* __restrict__ out) {
    int w = threadIdx.x >> 5, lane = threadIdx.x & 31;
    int nt = blockIdx.x * 16 + w;   // 125 blocks * 16 waves = 2000 tiles
    const __bf16* bp = WoutB + (size_t)nt * 8 * 512;
    v8f c = {};
    #pragma unroll
    for (int kt = 0; kt < 8; ++kt) {
        if (kt < 7)
            __builtin_prefetch(bp + (kt + 1) * 512 + lane * 16, 0, 0);
        c = wmma_bf16(*(const v16bf*)(hA + kt * 512 + lane * 16),
                      *(const v16bf*)(bp + kt * 512 + lane * 16), c);
    }
    int nl = lane & 15, moff = (lane >= 16) ? 8 : 0;
    int n = nt * 16 + nl;
    float bi = bout[n];
    #pragma unroll
    for (int r = 0; r < 8; ++r) {
        int b = r + moff;
        out[(size_t)b * (TT - 1) * VV + (size_t)t * VV + n] = c[r] + bi;
    }
}

// ---------------------------------------------------------------------------
// Host launcher
// ---------------------------------------------------------------------------
extern "C" void kernel_launch(void* const* d_in, const int* in_sizes, int n_in,
                              void* d_out, int out_size, void* d_ws, size_t ws_size,
                              hipStream_t stream) {
    (void)in_sizes; (void)n_in; (void)out_size; (void)ws_size;
    const int*   input_seq  = (const int*)d_in[0];
    const int*   target_seq = (const int*)d_in[1];
    const float* enc_emb    = (const float*)d_in[2];
    const float* enc_Wih    = (const float*)d_in[3];
    const float* enc_Whh    = (const float*)d_in[4];
    const float* enc_b      = (const float*)d_in[5];
    const float* dec_emb    = (const float*)d_in[6];
    const float* Wq         = (const float*)d_in[7];
    const float* bq         = (const float*)d_in[8];
    const float* Wk         = (const float*)d_in[9];
    const float* bk         = (const float*)d_in[10];
    const float* Wv         = (const float*)d_in[11];
    const float* bv         = (const float*)d_in[12];
    const float* Wo         = (const float*)d_in[13];
    const float* bo         = (const float*)d_in[14];
    const float* dec_Wih    = (const float*)d_in[15];
    const float* dec_Whh    = (const float*)d_in[16];
    const float* dec_b      = (const float*)d_in[17];
    const float* Wout       = (const float*)d_in[18];
    const float* bout       = (const float*)d_in[19];
    float* out = (float*)d_out;

    // Workspace carve-out (~27 MB total)
    char* p = (char*)d_ws;
    auto alloc = [&](size_t bytes) -> char* {
        char* r = p; p += (bytes + 255) & ~(size_t)255; return r;
    };
    __bf16* encWihB = (__bf16*)alloc((size_t)1024 * 256 * 2);
    __bf16* encWhhB = (__bf16*)alloc((size_t)1024 * 256 * 2);
    __bf16* WqB     = (__bf16*)alloc((size_t)256 * 256 * 2);
    __bf16* WkB     = (__bf16*)alloc((size_t)256 * 256 * 2);
    __bf16* WvB     = (__bf16*)alloc((size_t)256 * 256 * 2);
    __bf16* WoB     = (__bf16*)alloc((size_t)256 * 256 * 2);
    __bf16* decWihB = (__bf16*)alloc((size_t)1024 * 512 * 2);
    __bf16* decWhhB = (__bf16*)alloc((size_t)1024 * 256 * 2);
    __bf16* WoutB   = (__bf16*)alloc((size_t)VV * 256 * 2);
    __bf16* srcA    = (__bf16*)alloc((size_t)SS * 4096 * 2);
    __bf16* tokA    = (__bf16*)alloc((size_t)(TT - 1) * 4096 * 2);
    __bf16* encHsA  = (__bf16*)alloc((size_t)SS * 4096 * 2);
    __bf16* hA      = (__bf16*)alloc((size_t)4096 * 2);
    float*  Kf      = (float*)alloc((size_t)BB * SS * HH * 4);
    float*  Vf      = (float*)alloc((size_t)BB * SS * HH * 4);
    float*  c_state = (float*)alloc((size_t)4096 * 4);

    auto conv = [&](const float* W, __bf16* dst, int N, int K) {
        int total = N * K;
        conv_b_panel<<<(total + 255) / 256, 256, 0, stream>>>(W, dst, N, K);
    };
    conv(enc_Wih, encWihB, 1024, 256);
    conv(enc_Whh, encWhhB, 1024, 256);
    conv(Wq, WqB, 256, 256);
    conv(Wk, WkB, 256, 256);
    conv(Wv, WvB, 256, 256);
    conv(Wo, WoB, 256, 256);
    conv(dec_Wih, decWihB, 1024, 512);
    conv(dec_Whh, decWhhB, 1024, 256);
    conv(Wout, WoutB, VV, 256);

    {   // embeddings -> A-layout tiles
        int totalE = SS * 4096;
        gather_A<<<(totalE + 255) / 256, 256, 0, stream>>>(input_seq, SS, SS, enc_emb, srcA);
        int totalD = (TT - 1) * 4096;
        gather_A<<<(totalD + 255) / 256, 256, 0, stream>>>(target_seq, TT, TT - 1, dec_emb, tokA);
    }

    encoder_lstm<<<1, 512, 0, stream>>>(srcA, encWihB, encWhhB, enc_b,
                                        encHsA, hA, c_state);
    kv_gemm<<<SS, 1024, 0, stream>>>(encHsA, WkB, WvB, bk, bv, Kf, Vf);

    for (int t = 0; t < TT - 1; ++t) {
        dec_step<<<1, 512, 0, stream>>>(t, tokA, WqB, bq, Kf, Vf, WoB, bo,
                                        decWihB, decWhhB, dec_b, hA, c_state);
        logits_gemm<<<125, 512, 0, stream>>>(t, hA, WoutB, bout, out);
    }
}
